// Codebook_42099269435406
// MI455X (gfx1250) — matrix operand order repaired and checked by reading, not compiled, not run on hardware
//
#include <hip/hip_runtime.h>
#include <cmath>
#include <cstdint>

// ---------------------------------------------------------------------------
// VQ-VAE codebook bottleneck for MI455X (gfx1250), wave32 + bf16 WMMA.
//
// B=64, D=256, H=W=32 -> N=65536 tokens, DC=128, K=1024 codes.
//
// Roofline: minimum HBM traffic = x(64MB) + out(64MB) ~ 5.5us @ 23.3 TB/s.
// 21.5 GFLOP of GEMM work rides v_wmma_f32_16x16x32_bf16. Weights/codebook
// are pre-packed ONCE into the WMMA B-operand lane layout as bf16 (codebook
// pre-scaled by -2), hot loops are {async-to-LDS, ds_load_b128, v_wmma};
// f32->bf16 uses hardware v_cvt_pk_bf16_f32 when available.
//
//   K0a pack_preW   : pre_W  -> bf16 B-operand layout   [n][ks][lane][16]
//   K0b pack_lut    : -2*lut -> bf16 B-operand layout   [ct][ks][lane][16]
//   K1  proj_codebook: P[k,d] = lut[k,:].post_W[d,:] + post_b ; cnorm[k]
//   K2  fused_vq    : per wave 32 tokens:
//                      z = x.preW^T + preb (128 WMMA, B from LDS)
//                      z -> LDS -> A-operand bf16 (+ ||z||^2)
//                      4 codebook chunks staged in LDS (async-to-LDS +
//                      prefetch), 16 tiles/chunk, 8 WMMA/tile, running argmin
//   K3  gather_out  : out[b,d,h,w] = P[q,d]   (P = 1MB, L2-resident)
//   K4  finalize    : loss mean + perplexity
// ---------------------------------------------------------------------------

typedef __attribute__((ext_vector_type(16))) __bf16 v16bf;
typedef __attribute__((ext_vector_type(2)))  __bf16 v2bf;
typedef __attribute__((ext_vector_type(8)))  float  v8f;
typedef int v4i __attribute__((vector_size(16)));

#define N_TOKENS 65536
#define D_IN     256
#define DC_DIM   128
#define K_CODES  1024
#define HW_DIM   1024
#define OUT_ELEMS (64 * 256 * 32 * 32)   // 16777216

#if __has_builtin(__builtin_amdgcn_global_load_async_to_lds_b128)
#define HAVE_ASYNC_LDS 1
#else
#define HAVE_ASYNC_LDS 0
#endif

#if __has_builtin(__builtin_amdgcn_cvt_pk_bf16_f32)
#define HAVE_CVT_PK 1
#else
#define HAVE_CVT_PK 0
#endif

__device__ __forceinline__ unsigned short f2bf_bits(float f) {
    union { float f; unsigned u; } v; v.f = f;
    unsigned r = v.u + 0x7FFFu + ((v.u >> 16) & 1u);   // round-to-nearest-even
    return (unsigned short)(r >> 16);
}

// two floats -> packed bf16x2 (one v_cvt_pk_bf16_f32 when available)
__device__ __forceinline__ unsigned bf2_bits(float f0, float f1) {
#if HAVE_CVT_PK
    v2bf p = __builtin_amdgcn_cvt_pk_bf16_f32(f0, f1);
    unsigned u; __builtin_memcpy(&u, &p, 4);
    return u;
#else
    return (unsigned)f2bf_bits(f0) | ((unsigned)f2bf_bits(f1) << 16);
#endif
}

// write slots [s, s+1] of a v16bf A/B operand from two floats
__device__ __forceinline__ void bf_pair(v16bf& a, int s, float f0, float f1) {
#if HAVE_CVT_PK
    v2bf p = __builtin_amdgcn_cvt_pk_bf16_f32(f0, f1);
    a[s] = p[0]; a[s + 1] = p[1];
#else
    __bf16 h0, h1;
    unsigned short b0 = f2bf_bits(f0), b1 = f2bf_bits(f1);
    __builtin_memcpy(&h0, &b0, 2); __builtin_memcpy(&h1, &b1, 2);
    a[s] = h0; a[s + 1] = h1;
#endif
}

// A-operand (16x32 bf16) k index for slot s, half = lane>>4
// (pairs (s, s+1), s even, always map to consecutive k)
__device__ __forceinline__ int a_kidx(int s, int half) {
    return half * 16 + s;
}

// Block-cooperative 64KB global->LDS copy (128 threads, 512B each).
__device__ __forceinline__ void block_copy64k(void* dst, const void* src, int tid) {
    uint4* d = (uint4*)dst;
    const uint4* s = (const uint4*)src;
#if HAVE_ASYNC_LDS
    #pragma unroll
    for (int i = 0; i < 32; ++i) {
        __builtin_amdgcn_global_load_async_to_lds_b128(
            (__attribute__((address_space(1))) v4i*)(uintptr_t)(s + tid + 128 * i),
            (__attribute__((address_space(3))) v4i*)(d + tid + 128 * i),
            0, 0);
    }
#if __has_builtin(__builtin_amdgcn_s_wait_asynccnt)
    __builtin_amdgcn_s_wait_asynccnt(0);
#else
    asm volatile("s_wait_asynccnt 0" ::: "memory");
#endif
#else
    #pragma unroll
    for (int i = 0; i < 32; ++i) d[tid + 128 * i] = s[tid + 128 * i];
#endif
}

// ---------------------------------------------------------------------------
// K0a: pre_W [128,256] f32 -> PW bf16 in B-operand layout [n8][ks8][lane32][s16]
//      2 elements/thread (consecutive d), packed dword store.
// ---------------------------------------------------------------------------
__global__ void pack_preW_kernel(const float* __restrict__ preW,
                                 unsigned int* __restrict__ PW32) {
    const int pi  = blockIdx.x * 256 + threadIdx.x;       // 16384 pairs
    const int idx = pi * 2;
    const int s    = idx & 15;
    const int lane = (idx >> 4) & 31;
    const int ks   = (idx >> 9) & 7;
    const int n    = idx >> 12;
    const int dc = n * 16 + (lane & 15);
    const int d  = ks * 32 + (lane >> 4) * 16 + s;
    const float* p = preW + dc * D_IN + d;
    PW32[pi] = bf2_bits(p[0], p[1]);
}

// ---------------------------------------------------------------------------
// K0b: lookup [1024,128] f32 -> LB = bf16(-2*lut), B-operand layout
//      [ct64][ks4][lane32][s16]; 2 elements/thread (consecutive dc).
// ---------------------------------------------------------------------------
__global__ void pack_lut_kernel(const float* __restrict__ lut,
                                unsigned int* __restrict__ LB32) {
    const int pi  = blockIdx.x * 256 + threadIdx.x;       // 65536 pairs
    const int idx = pi * 2;
    const int s    = idx & 15;
    const int lane = (idx >> 4) & 31;
    const int ks   = (idx >> 9) & 3;
    const int ct   = idx >> 11;
    const int code = ct * 16 + (lane & 15);
    const int dc   = ks * 32 + (lane >> 4) * 16 + s;
    const float* p = lut + code * DC_DIM + dc;
    LB32[pi] = bf2_bits(-2.0f * p[0], -2.0f * p[1]);
}

// ---------------------------------------------------------------------------
// K1: projected codebook P[1024,256] (+post_b) and cnorm[1024]. 1024 waves.
// ---------------------------------------------------------------------------
__global__ void proj_codebook_kernel(const float* __restrict__ lut,
                                     const float* __restrict__ postW,
                                     const float* __restrict__ postb,
                                     float* __restrict__ P,
                                     float* __restrict__ cnorm) {
    const int lane  = threadIdx.x & 31;
    const int wave  = threadIdx.x >> 5;
    const int gwave = blockIdx.x * (blockDim.x >> 5) + wave;
    const int ct = gwave >> 4;
    const int dt = gwave & 15;
    const int m    = lane & 15;
    const int half = lane >> 4;

    v8f acc = {};
    float c2 = 0.0f;
    #pragma unroll
    for (int ks = 0; ks < 4; ++ks) {
        v16bf a, bm;
        #pragma unroll
        for (int s = 0; s < 16; s += 2) {
            const int dc = ks * 32 + a_kidx(s, half);
            const float v0 = lut[(ct * 16 + m) * DC_DIM + dc];
            const float v1 = lut[(ct * 16 + m) * DC_DIM + dc + 1];
            bf_pair(a, s, v0, v1);
            c2 += v0 * v0 + v1 * v1;
        }
        const float* brow = postW + (dt * 16 + m) * DC_DIM + ks * 32 + half * 16;
        #pragma unroll
        for (int s = 0; s < 16; s += 2) bf_pair(bm, s, brow[s], brow[s + 1]);
        acc = __builtin_amdgcn_wmma_f32_16x16x32_bf16(false, a, false, bm,
                                                      (short)0, acc, false, false);
    }
    c2 += __shfl_xor(c2, 16, 32);
    if (dt == 0 && lane < 16) cnorm[ct * 16 + m] = c2;

    #pragma unroll
    for (int j = 0; j < 8; ++j) {
        const int code = ct * 16 + j + half * 8;
        const int d    = dt * 16 + m;
        P[code * D_IN + d] = acc[j] + postb[d];
    }
}

// ---------------------------------------------------------------------------
// K2: fused pre-GEMM + quantization. Block = 4 waves; wave = 32 tokens
//     (two 16-token A tiles sharing every B operand -> 8 WMMA per B fetch).
// LDS: 64KB B-operand region (PW once, then 4 codebook chunks) +
//      32KB z relayout (8KB/wave) + 128 floats of ||z||^2.  ~98.8KB/block
//      -> 3 blocks (12 waves) per WGP against 320KB.
// ---------------------------------------------------------------------------
__global__ void fused_vq_kernel(const float* __restrict__ x,
                                const float* __restrict__ preb,
                                const unsigned short* __restrict__ PW,
                                const unsigned short* __restrict__ LB,
                                const float* __restrict__ cnorm,
                                float* __restrict__ qmap_out,
                                int*   __restrict__ q_ws,
                                float* __restrict__ counts,
                                float* __restrict__ loss_acc) {
    __shared__ __align__(16) unsigned char smemA[65536];   // B operands
    __shared__ __align__(16) unsigned char smemB[32768];   // z staging
    __shared__ float znorm_s[128];

    const int tid   = threadIdx.x;
    const int lane  = tid & 31;
    const int wave  = tid >> 5;
    const int m     = lane & 15;
    const int half  = lane >> 4;
    const int gwave = blockIdx.x * 4 + wave;
    const int token0 = gwave * 32;            // 32 tokens; 1024 % 32 == 0
    const int b   = token0 >> 10;
    const int hw0 = token0 & 1023;

    // stage packed pre_W (64KB) once per block
    block_copy64k(smemA, PW, tid);
    __syncthreads();

    // ---- Phase 1+2: per A-tile, z = x.preW^T, relayout to A operand -------
    v16bf za[8];                    // [tile0 ks0..3 | tile1 ks0..3]
    float* zrow = (float*)(smemB + wave * 8192);

    #pragma unroll
    for (int tt = 0; tt < 2; ++tt) {
        const int hwb = hw0 + tt * 16;

        v8f acc[8];
        #pragma unroll
        for (int n = 0; n < 8; ++n) acc[n] = (v8f){};

        for (int ks = 0; ks < 8; ++ks) {           // K = D = 256
            v16bf a;
            #pragma unroll
            for (int s = 0; s < 16; s += 2) {      // pairs: consecutive d
                const int d0 = ks * 32 + a_kidx(s, half);
                const float f0 = x[(b * D_IN + d0)     * HW_DIM + hwb + m];
                const float f1 = x[(b * D_IN + d0 + 1) * HW_DIM + hwb + m];
                bf_pair(a, s, f0, f1);
            }
            #pragma unroll
            for (int n = 0; n < 8; ++n) {
                v16bf bm;
                __builtin_memcpy(&bm, smemA + (((n * 8 + ks) * 32 + lane) << 5), 32);
                acc[n] = __builtin_amdgcn_wmma_f32_16x16x32_bf16(
                             false, a, false, bm, (short)0, acc[n], false, false);
            }
        }
        // C layout -> LDS [token][dc] (+ bias)
        #pragma unroll
        for (int n = 0; n < 8; ++n) {
            #pragma unroll
            for (int j = 0; j < 8; ++j) {
                const int tok = j + half * 8;
                const int dc  = n * 16 + m;
                zrow[tok * DC_DIM + dc] = acc[n][j] + preb[dc];
            }
        }
        // same-wave LDS ordering (DS ops are in-order; compiler adds waits)
        __builtin_amdgcn_fence(__ATOMIC_ACQ_REL, "workgroup");

        float z2 = 0.0f;
        #pragma unroll
        for (int ks = 0; ks < 4; ++ks) {
            #pragma unroll
            for (int s = 0; s < 16; s += 2) {      // pairs: consecutive dc
                const int dc = ks * 32 + a_kidx(s, half);
                const float v0 = zrow[m * DC_DIM + dc];
                const float v1 = zrow[m * DC_DIM + dc + 1];
                bf_pair(za[tt * 4 + ks], s, v0, v1);
                z2 += v0 * v0 + v1 * v1;
            }
        }
        z2 += __shfl_xor(z2, 16, 32);
        if (lane < 16) znorm_s[wave * 32 + tt * 16 + m] = z2;
    }

    // ---- Phase 3: stream codebook chunks through LDS, running argmin ------
    float bestv[2][8];
    int   besti[2][8];
    #pragma unroll
    for (int t = 0; t < 2; ++t)
        #pragma unroll
        for (int j = 0; j < 8; ++j) { bestv[t][j] = 3.0e38f; besti[t][j] = 0; }

    for (int c = 0; c < 4; ++c) {               // 4 chunks x 256 codes
        __syncthreads();                        // previous users done
        block_copy64k(smemA, LB + c * 32768, tid);
        __syncthreads();
        if (c < 3)                              // warm GL2 for the next chunk
            __builtin_prefetch(LB + (c + 1) * 32768 + tid * 256, 0, 1);

        for (int ctl = 0; ctl < 16; ++ctl) {
            v8f c0 = (v8f){}, c1 = (v8f){};
            #pragma unroll
            for (int ks = 0; ks < 4; ++ks) {
                v16bf bm;
                __builtin_memcpy(&bm, smemA + (((ctl * 4 + ks) * 32 + lane) << 5), 32);
                c0 = __builtin_amdgcn_wmma_f32_16x16x32_bf16(
                         false, za[ks],     false, bm, (short)0, c0, false, false);
                c1 = __builtin_amdgcn_wmma_f32_16x16x32_bf16(
                         false, za[4 + ks], false, bm, (short)0, c1, false, false);
            }
            const int   code = (c * 16 + ctl) * 16 + m;   // C column = lane%16
            const float cn   = cnorm[code];
            #pragma unroll
            for (int j = 0; j < 8; ++j) {                 // LB pre-scaled by -2
                const float d0 = cn + c0[j];
                if (d0 < bestv[0][j]) { bestv[0][j] = d0; besti[0][j] = code; }
                const float d1 = cn + c1[j];
                if (d1 < bestv[1][j]) { bestv[1][j] = d1; besti[1][j] = code; }
            }
        }
    }

    // cross-lane argmin over the 16 lanes of a token row (stay in half-wave)
    #pragma unroll
    for (int t = 0; t < 2; ++t) {
        #pragma unroll
        for (int j = 0; j < 8; ++j) {
            float v = bestv[t][j]; int i = besti[t][j];
            #pragma unroll
            for (int mask = 1; mask <= 8; mask <<= 1) {
                const float ov = __shfl_xor(v, mask, 32);
                const int   oi = __shfl_xor(i, mask, 32);
                if (ov < v || (ov == v && oi < i)) { v = ov; i = oi; }
            }
            bestv[t][j] = v; besti[t][j] = i;
        }
    }

    if (m == 0) {                 // lanes 0 and 16 each finalize 16 tokens
        float lsum = 0.0f;
        #pragma unroll
        for (int t = 0; t < 2; ++t) {
            #pragma unroll
            for (int j = 0; j < 8; ++j) {
                const int tl    = t * 16 + j + half * 8;
                const int token = token0 + tl;
                q_ws[token]     = besti[t][j];
                qmap_out[token] = (float)besti[t][j];
                atomicAdd(&counts[besti[t][j]], 1.0f);
                lsum += znorm_s[wave * 32 + tl] + bestv[t][j];  // ||z - c_q||^2
            }
        }
        atomicAdd(loss_acc, lsum);
    }
}

// ---------------------------------------------------------------------------
// K3: out[b,d,h,w] = P[q[b,h,w], d]
// ---------------------------------------------------------------------------
__global__ void gather_out_kernel(const float* __restrict__ P,
                                  const int* __restrict__ q_ws,
                                  float* __restrict__ out) {
    const int idx = blockIdx.x * 256 + threadIdx.x;
    const int hw  = idx & 1023;
    const int d   = (idx >> 10) & 255;
    const int b   = idx >> 18;
    out[idx] = P[q_ws[(b << 10) | hw] * D_IN + d];
}

// ---------------------------------------------------------------------------
// K4: loss mean + perplexity. One block, 1024 threads.
// ---------------------------------------------------------------------------
__global__ void finalize_kernel(const float* __restrict__ counts,
                                const float* __restrict__ loss_acc,
                                float* __restrict__ loss_out,
                                float* __restrict__ perp_out) {
    __shared__ float red[32];
    const int t    = threadIdx.x;
    const int lane = t & 31;
    const float e  = counts[t] * (1.0f / (float)N_TOKENS);
    float term = e * logf(e + 1e-10f);
    #pragma unroll
    for (int mask = 1; mask <= 16; mask <<= 1) term += __shfl_xor(term, mask, 32);
    if (lane == 0) red[t >> 5] = term;
    __syncthreads();
    if (t < 32) {
        float v = red[t];
        #pragma unroll
        for (int mask = 1; mask <= 16; mask <<= 1) v += __shfl_xor(v, mask, 32);
        if (t == 0) {
            *perp_out = expf(-v);
            *loss_out = *loss_acc * (1.0f / ((float)N_TOKENS * (float)DC_DIM));
        }
    }
}

// ---------------------------------------------------------------------------
extern "C" void kernel_launch(void* const* d_in, const int* in_sizes, int n_in,
                              void* d_out, int out_size, void* d_ws, size_t ws_size,
                              hipStream_t stream) {
    (void)in_sizes; (void)n_in; (void)out_size; (void)ws_size;

    const float* x     = (const float*)d_in[0];
    const float* preW  = (const float*)d_in[1];
    const float* preb  = (const float*)d_in[2];
    const float* postW = (const float*)d_in[3];
    const float* postb = (const float*)d_in[4];
    const float* lut   = (const float*)d_in[5];

    float* out      = (float*)d_out;                       // [64,256,32,32]
    float* loss_out = out + OUT_ELEMS;                     // scalar
    float* qmap     = out + OUT_ELEMS + 1;                 // [64,32,32]
    float* perp_out = out + OUT_ELEMS + 1 + N_TOKENS;      // scalar

    // Workspace layout (16B-aligned sections)
    float* P        = (float*)d_ws;                        // 262144 f
    float* cnorm    = P + K_CODES * D_IN;                  // 1024 f
    float* counts   = cnorm + K_CODES;                     // 1024 f
    float* loss_acc = counts + K_CODES;                    // 1 f (+3 pad)
    int*   q_ws     = (int*)(loss_acc + 4);                // 65536 i32
    unsigned short* LB = (unsigned short*)(q_ws + N_TOKENS); // 131072 u16 (256KB)
    unsigned short* PW = LB + 131072;                        // 32768 u16 (64KB)

    (void)hipMemsetAsync(counts, 0, (K_CODES + 4) * sizeof(float), stream);

    pack_preW_kernel<<<64, 256, 0, stream>>>(preW, (unsigned int*)PW);
    pack_lut_kernel<<<256, 256, 0, stream>>>(lut, (unsigned int*)LB);
    proj_codebook_kernel<<<256, 128, 0, stream>>>(lut, postW, postb, P, cnorm);
    fused_vq_kernel<<<N_TOKENS / 128, 128, 0, stream>>>(x, preb, PW, LB, cnorm,
                                                        qmap, q_ws, counts, loss_acc);
    gather_out_kernel<<<OUT_ELEMS / 256, 256, 0, stream>>>(P, q_ws, out);
    finalize_kernel<<<1, 1024, 0, stream>>>(counts, loss_acc, loss_out, perp_out);
}
